// RectifiedLinearAttention_65438121721931
// MI455X (gfx1250) — matrix operand order repaired and checked by reading, not compiled, not run on hardware
//
#include <hip/hip_runtime.h>

typedef __attribute__((ext_vector_type(16))) __bf16 v16bf;
typedef __attribute__((ext_vector_type(8)))  float  v8f;

#define B_   2
#define S_   2048
#define D_   1024
#define H_   16
#define DH_  64
#define M_   (B_ * S_)   /* 4096 token rows */
#define N3_  (3 * D_)    /* 3072 qkv cols   */
#define LDT  40          /* LDS tile stride in bf16: 80B = 16B multiple, bank-friendly */

// ---------- helpers ----------

__device__ __forceinline__ __bf16 f2bf(float f) {
  unsigned u = __builtin_bit_cast(unsigned, f);
  u += 0x7FFFu + ((u >> 16) & 1u);           // round-to-nearest-even
  return __builtin_bit_cast(__bf16, (unsigned short)(u >> 16));
}

// two floats -> packed bf16x2 in a dword (lo = a, hi = b)
__device__ __forceinline__ unsigned pack2bf(float a, float b) {
  unsigned ua = __builtin_bit_cast(unsigned, a);
  ua += 0x7FFFu + ((ua >> 16) & 1u);
  unsigned ub = __builtin_bit_cast(unsigned, b);
  ub += 0x7FFFu + ((ub >> 16) & 1u);
  return (ua >> 16) | (ub & 0xFFFF0000u);
}

__device__ __forceinline__ v8f wmma_bf16(v16bf a, v16bf b, v8f c) {
  // D(16x16,f32) = A(16x32,bf16) x B(32x16,bf16) + C
  return __builtin_amdgcn_wmma_f32_16x16x32_bf16(
      /*neg_a=*/false, a, /*neg_b=*/false, b,
      /*c_mod=*/(short)0, c, /*reuse_a=*/false, /*reuse_b=*/false);
}

union frag_u { uint4 u[2]; v16bf v; };

// A-fragment (16x32 bf16) from a row-major tile, row stride LDT (16B-aligned rows).
// Contiguous K order (same as ldsBt_frag): lanes 0-15 row=lane, K=0..15; lanes 16-31 row=lane-16,
// K=16..31. 16 contiguous K-elements -> 2x b128 loads.
__device__ __forceinline__ v16bf ldsA_frag(const __bf16* a) {
  const int lane = threadIdx.x & 31;
  const int row  = lane & 15;
  const int kb   = (lane >> 4) << 4;
  const __bf16* p = a + row * LDT + kb;
  frag_u t;
  t.u[0] = *(const uint4*)(p);
  t.u[1] = *(const uint4*)(p + 8);
  return t.v;
}

// B-fragment (32x16 bf16) from an N-MAJOR (transposed) tile: bt[n][k], row
// stride LDT. lanes 0-15 col=lane K=0..15; lanes 16-31 col=lane-16 K=16..31.
// 16 contiguous K-elements -> 2x b128 loads.
__device__ __forceinline__ v16bf ldsBt_frag(const __bf16* bt) {
  const int lane = threadIdx.x & 31;
  const int n    = lane & 15;
  const int kb   = (lane >> 4) << 4;
  const __bf16* p = bt + n * LDT + kb;
  frag_u t;
  t.u[0] = *(const uint4*)(p);
  t.u[1] = *(const uint4*)(p + 8);
  return t.v;
}

// ---------- kernel 1: QKV projection (x fp32 -> bf16 WMMA -> head-split Q/K/V) ----------
// 128x128 tile per workgroup, 8 waves in 4x2 grid, 32x64 per wave.

__global__ __launch_bounds__(256) void qkv_proj_kernel(
    const float* __restrict__ x, const float* __restrict__ Wqkv,
    __bf16* __restrict__ qw, __bf16* __restrict__ kw, __bf16* __restrict__ vw)
{
  __shared__ alignas(16) __bf16 As[128][LDT];   // 128 rows x 32 k
  __shared__ alignas(16) __bf16 Bt[128][LDT];   // 128 n   x 32 k (transposed)

  const int tid  = threadIdx.x;
  const int lane = tid & 31;
  const int w    = tid >> 5;
  const int wm   = w >> 1, wn = w & 1;
  const int mBase = blockIdx.y * 128;
  const int nBase = blockIdx.x * 128;

  v8f acc0[4] = {{}, {}, {}, {}};
  v8f acc1[4] = {{}, {}, {}, {}};

  for (int k0 = 0; k0 < D_; k0 += 32) {
    // stage A: 128x32 fp32 -> bf16, packed b32 stores
    for (int i = tid; i < (128 * 32) / 4; i += 256) {
      const int lin = i << 2;
      const int row = lin >> 5;
      const int kc  = lin & 31;
      const float4 f = *(const float4*)(x + (size_t)(mBase + row) * D_ + k0 + kc);
      *(unsigned*)&As[row][kc + 0] = pack2bf(f.x, f.y);
      *(unsigned*)&As[row][kc + 2] = pack2bf(f.z, f.w);
    }
    // stage B transposed: Wqkv[k][n] -> Bt[n][k]
    for (int i = tid; i < (32 * 128) / 4; i += 256) {
      const int lin = i << 2;
      const int kr  = lin >> 7;
      const int nc  = lin & 127;
      const float4 f = *(const float4*)(Wqkv + (size_t)(k0 + kr) * N3_ + nBase + nc);
      Bt[nc + 0][kr] = f2bf(f.x);
      Bt[nc + 1][kr] = f2bf(f.y);
      Bt[nc + 2][kr] = f2bf(f.z);
      Bt[nc + 3][kr] = f2bf(f.w);
    }
    if (k0 + 32 < D_)  // gfx1250 global_prefetch_b8 of next A k-slab
      __builtin_prefetch(x + (size_t)(mBase + (tid >> 1)) * D_ + k0 + 32, 0, 0);
    __syncthreads();

    const v16bf a0 = ldsA_frag(&As[wm * 32 +  0][0]);
    const v16bf a1 = ldsA_frag(&As[wm * 32 + 16][0]);
    v16bf bf[4];
#pragma unroll
    for (int j = 0; j < 4; ++j) bf[j] = ldsBt_frag(&Bt[wn * 64 + j * 16][0]);
#pragma unroll
    for (int j = 0; j < 4; ++j) {
      acc0[j] = wmma_bf16(a0, bf[j], acc0[j]);
      acc1[j] = wmma_bf16(a1, bf[j], acc1[j]);
    }
    __syncthreads();
  }

  // epilogue: scatter into head-split [BH, S, DH] bf16; Q scaled by dh^-0.5
  const int laneN = lane & 15;
  const int rBase = (lane >> 4) << 3;
#pragma unroll
  for (int i = 0; i < 2; ++i)
#pragma unroll
    for (int j = 0; j < 4; ++j) {
      const v8f acc = i == 0 ? acc0[j] : acc1[j];
      const int colg = nBase + wn * 64 + j * 16 + laneN;
      const int sel  = colg >> 10;         // 0:q 1:k 2:v
      const int rem  = colg & 1023;
      const int h    = rem >> 6, c = rem & 63;
      __bf16* dst = (sel == 0) ? qw : ((sel == 1) ? kw : vw);
      const float mult = (sel == 0) ? 0.125f : 1.0f;   // dh^-0.5 = 1/8
#pragma unroll
      for (int r = 0; r < 8; ++r) {
        const int rowg = mBase + wm * 32 + i * 16 + rBase + r;
        const int b = rowg >> 11, s = rowg & 2047;
        dst[((size_t)((b * H_ + h) * S_ + s)) * DH_ + c] = f2bf(acc[r] * mult);
      }
    }
}

// ---------- kernel 2: causal ReLU attention O = relu(mask o (QK^T)) V ----------

__global__ __launch_bounds__(128) void relu_attn_kernel(
    const __bf16* __restrict__ qw, const __bf16* __restrict__ kw,
    const __bf16* __restrict__ vw, float* __restrict__ ow)
{
  __shared__ alignas(16) __bf16 Vt[64][LDT];       // chan-major V tile (64 chans x 32 keys)
  __shared__ alignas(16) __bf16 Sst[4][16][LDT];   // per-wave relu(S) staging

  const int tid  = threadIdx.x;
  const int lane = tid & 31;
  const int w    = tid >> 5;          // 4 waves x 16 q-rows = 64 q-rows/block
  const int bh   = blockIdx.y;
  const int b    = bh >> 4, h = bh & 15;
  const int qbase = blockIdx.x * 64;

  const size_t hoff = (size_t)bh * S_ * DH_;
  const __bf16* Q  = qw + hoff;
  const __bf16* Kp = kw + hoff;
  const __bf16* Vp = vw + hoff;

  const int laneN = lane & 15;
  const int rBase = (lane >> 4) << 3;  // C-layout row base
  const int akb   = (lane >> 4) << 4;  // A-frag K base (elements)
  const int bkb   = (lane >> 4) << 4;  // B-frag K base (elements)

  // Q fragments: A(16 x dh=64) as two K=32 fragments, b128 global loads
  v16bf aq0, aq1;
  {
    const int qrow = qbase + w * 16 + (lane & 15);
    const __bf16* qp = Q + (size_t)qrow * DH_;
    frag_u t0, t1;
    t0.u[0] = *(const uint4*)(qp + akb);
    t0.u[1] = *(const uint4*)(qp + 8 + akb);
    t1.u[0] = *(const uint4*)(qp + 32 + akb);
    t1.u[1] = *(const uint4*)(qp + 40 + akb);
    aq0 = t0.v; aq1 = t1.v;
  }

  v8f accO[4] = {{}, {}, {}, {}};

  // causal: only key tiles with kb <= qbase+63 contribute
  for (int kb = 0; kb < qbase + 64; kb += 32) {
    __syncthreads();   // protect Vt from previous iteration's readers
    // stage V transposed: Vp[key][chan] -> Vt[chan][key], b128 global loads
    for (int i = tid; i < (32 * 64) / 8; i += 128) {
      const int key = i >> 3;
      const int c8  = (i & 7) << 3;
      union { uint4 u; __bf16 e[8]; } vv;
      vv.u = *(const uint4*)(Vp + (size_t)(kb + key) * DH_ + c8);
#pragma unroll
      for (int t = 0; t < 8; ++t) Vt[c8 + t][key] = vv.e[t];
    }
    __syncthreads();

#pragma unroll
    for (int t = 0; t < 2; ++t) {
      // B = K^T (chan x key): lane holds key col, chans contiguous -> b128 loads
      const __bf16* kp = Kp + (size_t)(kb + t * 16 + laneN) * DH_;
      frag_u k0, k1;
      k0.u[0] = *(const uint4*)(kp + bkb);
      k0.u[1] = *(const uint4*)(kp + bkb + 8);
      k1.u[0] = *(const uint4*)(kp + 32 + bkb);
      k1.u[1] = *(const uint4*)(kp + 32 + bkb + 8);

      v8f sacc = {};
      sacc = wmma_bf16(aq0, k0.v, sacc);
      sacc = wmma_bf16(aq1, k1.v, sacc);

      // causal-zero + relu, stage to LDS (C layout -> row-major bf16)
#pragma unroll
      for (int r = 0; r < 8; ++r) {
        const int qg = qbase + w * 16 + rBase + r;
        const int kg = kb + t * 16 + laneN;
        float sv = sacc[r];
        sv = (kg <= qg) ? fmaxf(sv, 0.0f) : 0.0f;
        Sst[w][rBase + r][t * 16 + laneN] = f2bf(sv);
      }
    }
    // relu(S) as A(16x32); V fragments from transposed LDS tile
    const v16bf sa = ldsA_frag(&Sst[w][0][0]);
#pragma unroll
    for (int nb = 0; nb < 4; ++nb) {
      const v16bf bv = ldsBt_frag(&Vt[nb * 16][0]);
      accO[nb] = wmma_bf16(sa, bv, accO[nb]);
    }
  }

  // combine heads: O[b, s, h*64 + c] fp32
#pragma unroll
  for (int nb = 0; nb < 4; ++nb) {
    const int d = h * DH_ + nb * 16 + laneN;
#pragma unroll
    for (int r = 0; r < 8; ++r) {
      const int qg = qbase + w * 16 + rBase + r;
      ow[((size_t)(b * S_ + qg)) * D_ + d] = accO[nb][r];
    }
  }
}

// ---------- kernel 3: gated RMSNorm (fp32 in -> bf16 out) ----------

__global__ __launch_bounds__(256) void rmsnorm_gate_kernel(
    const float* __restrict__ ow, const float* __restrict__ scale,
    const float* __restrict__ gate, __bf16* __restrict__ yw)
{
  __shared__ float red[256];
  const int row = blockIdx.x;
  const int tid = threadIdx.x;
  const float4 v4 = ((const float4*)(ow + (size_t)row * D_))[tid];

  red[tid] = v4.x * v4.x + v4.y * v4.y + v4.z * v4.z + v4.w * v4.w;
  __syncthreads();
  for (int s = 128; s > 0; s >>= 1) {
    if (tid < s) red[tid] += red[tid + s];
    __syncthreads();
  }
  const float rinv = rsqrtf(red[0] * (1.0f / D_) + 1e-7f);
  const float4 g4 = ((const float4*)gate)[tid];
  const float4 s4 = ((const float4*)scale)[tid];
  float y0 = s4.x * v4.x * rinv / (1.0f + __expf(-g4.x * v4.x));
  float y1 = s4.y * v4.y * rinv / (1.0f + __expf(-g4.y * v4.y));
  float y2 = s4.z * v4.z * rinv / (1.0f + __expf(-g4.z * v4.z));
  float y3 = s4.w * v4.w * rinv / (1.0f + __expf(-g4.w * v4.w));
  uint2 packed;
  packed.x = pack2bf(y0, y1);
  packed.y = pack2bf(y2, y3);
  *(uint2*)(yw + (size_t)row * D_ + tid * 4) = packed;
}

// ---------- kernel 4: output projection + bias (bf16 WMMA -> fp32) ----------

__global__ __launch_bounds__(256) void out_proj_kernel(
    const __bf16* __restrict__ yw, const float* __restrict__ W_out,
    const float* __restrict__ b_out, float* __restrict__ out)
{
  __shared__ alignas(16) __bf16 As[128][LDT];
  __shared__ alignas(16) __bf16 Bt[128][LDT];

  const int tid  = threadIdx.x;
  const int lane = tid & 31;
  const int w    = tid >> 5;
  const int wm   = w >> 1, wn = w & 1;
  const int mBase = blockIdx.y * 128;
  const int nBase = blockIdx.x * 128;

  v8f acc0[4] = {{}, {}, {}, {}};
  v8f acc1[4] = {{}, {}, {}, {}};

  for (int k0 = 0; k0 < D_; k0 += 32) {
    // stage A (already bf16): b128 copy, 8 elements per op
    for (int i = tid; i < (128 * 32) / 8; i += 256) {
      const int row = i >> 2;
      const int kc  = (i & 3) << 3;
      *(uint4*)&As[row][kc] = *(const uint4*)(yw + (size_t)(mBase + row) * D_ + k0 + kc);
    }
    // stage B transposed: W_out[k][n] fp32 -> Bt[n][k] bf16
    for (int i = tid; i < (32 * 128) / 4; i += 256) {
      const int lin = i << 2;
      const int kr  = lin >> 7;
      const int nc  = lin & 127;
      const float4 f = *(const float4*)(W_out + (size_t)(k0 + kr) * D_ + nBase + nc);
      Bt[nc + 0][kr] = f2bf(f.x);
      Bt[nc + 1][kr] = f2bf(f.y);
      Bt[nc + 2][kr] = f2bf(f.z);
      Bt[nc + 3][kr] = f2bf(f.w);
    }
    __syncthreads();

    const v16bf a0 = ldsA_frag(&As[wm * 32 +  0][0]);
    const v16bf a1 = ldsA_frag(&As[wm * 32 + 16][0]);
    v16bf bf[4];
#pragma unroll
    for (int j = 0; j < 4; ++j) bf[j] = ldsBt_frag(&Bt[wn * 64 + j * 16][0]);
#pragma unroll
    for (int j = 0; j < 4; ++j) {
      acc0[j] = wmma_bf16(a0, bf[j], acc0[j]);
      acc1[j] = wmma_bf16(a1, bf[j], acc1[j]);
    }
    __syncthreads();
  }

  const int laneN = lane & 15;
  const int rBase = (lane >> 4) << 3;
#pragma unroll
  for (int i = 0; i < 2; ++i)
#pragma unroll
    for (int j = 0; j < 4; ++j) {
      const v8f acc = i == 0 ? acc0[j] : acc1[j];
      const int colg = nBase + wn * 64 + j * 16 + laneN;
      const float bias = b_out[colg];
#pragma unroll
      for (int r = 0; r < 8; ++r) {
        const int rowg = mBase + wm * 32 + i * 16 + rBase + r;
        out[(size_t)rowg * D_ + colg] = acc[r] + bias;
      }
    }
}

// ---------- launch ----------

extern "C" void kernel_launch(void* const* d_in, const int* in_sizes, int n_in,
                              void* d_out, int out_size, void* d_ws, size_t ws_size,
                              hipStream_t stream) {
  const float* x     = (const float*)d_in[0];
  // d_in[1] = mem_mask: causal; applied analytically inside relu_attn_kernel
  const float* Wqkv  = (const float*)d_in[2];
  const float* W_out = (const float*)d_in[3];
  const float* b_out = (const float*)d_in[4];
  const float* scale = (const float*)d_in[5];
  const float* gate  = (const float*)d_in[6];
  float* out = (float*)d_out;

  const size_t nTok = (size_t)M_ * D_;      // 4M elements
  __bf16* q_ws = (__bf16*)d_ws;             //  8 MB
  __bf16* k_ws = q_ws + nTok;               //  8 MB
  __bf16* v_ws = k_ws + nTok;               //  8 MB
  float*  o_ws = (float*)(v_ws + nTok);     // 16 MB
  __bf16* y_ws = (__bf16*)(o_ws + nTok);    //  8 MB   (total 48 MB of d_ws)

  qkv_proj_kernel<<<dim3(N3_ / 128, M_ / 128), 256, 0, stream>>>(x, Wqkv, q_ws, k_ws, v_ws);
  relu_attn_kernel<<<dim3(S_ / 64, B_ * H_), 128, 0, stream>>>(q_ws, k_ws, v_ws, o_ws);
  rmsnorm_gate_kernel<<<M_, 256, 0, stream>>>(o_ws, scale, gate, y_ws);
  out_proj_kernel<<<dim3(D_ / 128, M_ / 128), 256, 0, stream>>>(y_ws, W_out, b_out, out);
}
